// MultiHeadMetaGatedTitansLayer_18743237279812
// MI455X (gfx1250) — matrix operand order repaired and checked by reading, not compiled, not run on hardware
//
#include <hip/hip_runtime.h>
#include <hip/hip_bf16.h>
#include <math.h>

#define BB   4096
#define DD   512
#define HH   8
#define DHH  64
#define NMETA 1040   // 2*D + 2*H

typedef __attribute__((ext_vector_type(16))) _Float16 v16h;
typedef __attribute__((ext_vector_type(8)))  _Float16 v8h;
typedef __attribute__((ext_vector_type(8)))  float    v8f;

static __device__ __forceinline__ float sigmoidf_(float x) { return 1.f / (1.f + __expf(-x)); }

// ---------------------------------------------------------------------------
// Block-wide sum over 256 threads (8 wave32s). `red` is __shared__ float[8].
// ---------------------------------------------------------------------------
static __device__ __forceinline__ float block_sum256(float v, float* red) {
#pragma unroll
    for (int off = 16; off; off >>= 1) v += __shfl_xor(v, off, 32);
    const int wid = threadIdx.x >> 5;
    if ((threadIdx.x & 31) == 0) red[wid] = v;
    __syncthreads();
    if (threadIdx.x == 0) {
        float s = 0.f;
#pragma unroll
        for (int i = 0; i < 8; ++i) s += red[i];
        red[0] = s;
    }
    __syncthreads();
    float r = red[0];
    __syncthreads();   // allow red[] reuse by caller
    return r;
}

// ---------------------------------------------------------------------------
// f32 -> f16 conversion (weights)
// ---------------------------------------------------------------------------
__global__ void cvt_f16(const float* __restrict__ src, _Float16* __restrict__ dst, int n) {
    int i = blockIdx.x * blockDim.x + threadIdx.x;
    if (i < n) dst[i] = (_Float16)src[i];
}

// ---------------------------------------------------------------------------
// WMMA GEMM:  C[M,N] = A[M,K] @ W[N,K]^T   (A,W f16 row-major, C f32)
// One 16x16 tile per wave32, K-loop of V_WMMA_F32_16X16X32_F16.
// Fragment layouts: contiguous K (k = 16*hi + s) in both A and B.
// ---------------------------------------------------------------------------
__global__ void wmma_gemm_nt(const _Float16* __restrict__ A,
                             const _Float16* __restrict__ W,
                             float* __restrict__ C,
                             int M, int N, int K)
{
    const int wave = (blockIdx.x * blockDim.x + threadIdx.x) >> 5;
    const int lane = threadIdx.x & 31;
    const int nt   = N >> 4;
    const int tm   = wave / nt;
    const int tn   = wave - tm * nt;
    const int hi   = lane >> 4;          // lane group (0: lanes 0-15, 1: lanes 16-31)
    const int l15  = lane & 15;

    // A: lane holds row (tm*16+l15); 16 contiguous K, +16 for hi lanes
    const _Float16* Ap = A + (size_t)(tm * 16 + l15) * K + (hi ? 16 : 0);
    // B: lane holds column (tn*16+l15) of W^T = row of W; 16 contiguous K, +16 for hi lanes
    const _Float16* Wp = W + (size_t)(tn * 16 + l15) * K + (hi ? 16 : 0);

    v8f acc = {};
    for (int k0 = 0; k0 < K; k0 += 32) {
        v8h a0 = *(const v8h*)(Ap + k0);
        v8h a1 = *(const v8h*)(Ap + k0 + 8);
        v16h a;
#pragma unroll
        for (int e = 0; e < 8; ++e) { a[e] = a0[e]; a[e + 8] = a1[e]; }
        v16h b = *(const v16h*)(Wp + k0);
        acc = __builtin_amdgcn_wmma_f32_16x16x32_f16(
            /*neg_a=*/false, a, /*neg_b=*/false, b,
            /*c_mod=*/(short)0, acc, /*reuse_a=*/false, /*reuse_b=*/false);
    }
    // C/D layout: VGPR r -> M = tm*16 + r + 8*hi, N = tn*16 + l15
    float* Cp = C + (size_t)(tm * 16 + hi * 8) * N + tn * 16 + l15;
#pragma unroll
    for (int r = 0; r < 8; ++r) Cp[(size_t)r * N] = acc[r];
}

// ---------------------------------------------------------------------------
// LayerNorm over D=512 for item (rows [0,B)) and user_static (rows [B,2B)).
// Item rows -> i_norm f32 + i_norm f16 (GEMM A operand).
// User rows -> first half of meta_in (f16), stride 1024.
// ---------------------------------------------------------------------------
__global__ void ln_dual(const float* __restrict__ item,
                        const float* __restrict__ user,
                        const float* __restrict__ g, const float* __restrict__ b,
                        float* __restrict__ i_norm, _Float16* __restrict__ i_norm_h,
                        _Float16* __restrict__ meta_in_h)
{
    __shared__ float red[8];
    const int t = threadIdx.x;
    const int row = blockIdx.x;
    const bool isItem = row < BB;
    const float* src = isItem ? (item + (size_t)row * DD)
                              : (user + (size_t)(row - BB) * DD);
    float x0 = src[t], x1 = src[t + 256];
    float m   = block_sum256(x0 + x1, red) * (1.f / DD);
    float d0 = x0 - m, d1 = x1 - m;
    float var = block_sum256(d0 * d0 + d1 * d1, red) * (1.f / DD);
    float inv = rsqrtf(var + 1e-5f);
    float y0 = d0 * inv * g[t]       + b[t];
    float y1 = d1 * inv * g[t + 256] + b[t + 256];
    if (isItem) {
        size_t o = (size_t)row * DD;
        i_norm[o + t] = y0;           i_norm[o + t + 256] = y1;
        i_norm_h[o + t] = (_Float16)y0; i_norm_h[o + t + 256] = (_Float16)y1;
    } else {
        size_t o = (size_t)(row - BB) * 1024;
        meta_in_h[o + t] = (_Float16)y0; meta_in_h[o + t + 256] = (_Float16)y1;
    }
}

// ---------------------------------------------------------------------------
// Per (b): for each head h, l2-normalize q[b,h,:], then
// memory_context[b,h,i] = sum_j old_state[b,h,i,j]*q[b,h,j].
// Writes f16 into right half of meta_in. State tile staged in LDS (16 KB).
// ---------------------------------------------------------------------------
__global__ void qnorm_memctx(const float* __restrict__ q_raw,
                             const float* __restrict__ old_state,
                             _Float16* __restrict__ meta_in_h)
{
    __shared__ float sS[4096];
    __shared__ float qn[64];
    __shared__ float pm[256];
    const int t = threadIdx.x;
    const int b = blockIdx.x;
    for (int h = 0; h < HH; ++h) {
        const size_t bh = (size_t)b * HH + h;
        const float* sg = old_state + bh * 4096;
        for (int idx = t; idx < 4096; idx += 256) sS[idx] = sg[idx];
        float qv = (t < 64) ? q_raw[(size_t)b * DD + h * 64 + t] : 0.f;
        if (t < 64) pm[t] = qv * qv;
        __syncthreads();
        for (int s2 = 32; s2; s2 >>= 1) { if (t < s2) pm[t] += pm[t + s2]; __syncthreads(); }
        float inv = 1.f / fmaxf(sqrtf(pm[0]), 1e-12f);
        __syncthreads();
        if (t < 64) qn[t] = qv * inv;
        __syncthreads();
        {   // 4-way split row dot
            int i = t >> 2, seg = t & 3;
            float p = 0.f;
            const float* rowp = &sS[i * 64 + seg * 16];
            const float* qp = &qn[seg * 16];
#pragma unroll
            for (int j = 0; j < 16; ++j) p += rowp[j] * qp[j];
            pm[t] = p;
        }
        __syncthreads();
        if (t < 64) {
            float mc = pm[t * 4] + pm[t * 4 + 1] + pm[t * 4 + 2] + pm[t * 4 + 3];
            meta_in_h[(size_t)b * 1024 + 512 + h * 64 + t] = (_Float16)mc;
        }
        __syncthreads();
    }
}

// ---------------------------------------------------------------------------
// hmid = relu(LN(hraw + bm1, gml, bml)) -> f16
// ---------------------------------------------------------------------------
__global__ void ln_relu(const float* __restrict__ hraw, const float* __restrict__ bm1,
                        const float* __restrict__ gml, const float* __restrict__ bml,
                        _Float16* __restrict__ hmid_h)
{
    __shared__ float red[8];
    const int t = threadIdx.x, row = blockIdx.x;
    const float* src = hraw + (size_t)row * DD;
    float x0 = src[t] + bm1[t], x1 = src[t + 256] + bm1[t + 256];
    float m   = block_sum256(x0 + x1, red) * (1.f / DD);
    float d0 = x0 - m, d1 = x1 - m;
    float var = block_sum256(d0 * d0 + d1 * d1, red) * (1.f / DD);
    float inv = rsqrtf(var + 1e-5f);
    float y0 = fmaxf(d0 * inv * gml[t] + bml[t], 0.f);
    float y1 = fmaxf(d1 * inv * gml[t + 256] + bml[t + 256], 0.f);
    size_t o = (size_t)row * DD;
    hmid_h[o + t] = (_Float16)y0;
    hmid_h[o + t + 256] = (_Float16)y1;
}

// ---------------------------------------------------------------------------
// modulated = i_norm * (1 + tanh(gamma)) + beta   (gamma/beta from meta_out+bm2)
// ---------------------------------------------------------------------------
__global__ void modulate(const float* __restrict__ meta_raw, const float* __restrict__ bm2,
                         const float* __restrict__ i_norm, _Float16* __restrict__ mod_h)
{
    size_t idx = (size_t)blockIdx.x * blockDim.x + threadIdx.x; // over B*D
    int b = (int)(idx >> 9);
    int d = (int)(idx & 511);
    float gamma = meta_raw[(size_t)b * NMETA + d]       + bm2[d];
    float beta  = meta_raw[(size_t)b * NMETA + 512 + d] + bm2[512 + d];
    float y = i_norm[idx] * (1.f + tanhf(gamma)) + beta;
    mod_h[idx] = (_Float16)y;
}

// ---------------------------------------------------------------------------
// alpha/eta gates: 8 waves per row b, wave h does two 512-dots + sigmoid.
// ---------------------------------------------------------------------------
__global__ void ae_kernel(const _Float16* __restrict__ mod_h,
                          const float* __restrict__ Wa, const float* __restrict__ ba,
                          const float* __restrict__ We, const float* __restrict__ be,
                          const float* __restrict__ meta_raw, const float* __restrict__ bm2,
                          float* __restrict__ alpha, float* __restrict__ eta)
{
    const int b = blockIdx.x;
    const int lane = threadIdx.x & 31;
    const int h = threadIdx.x >> 5;
    const _Float16* mrow = mod_h + (size_t)b * DD;
    float sa = 0.f, se = 0.f;
    for (int d = lane; d < DD; d += 32) {
        float mv = (float)mrow[d];
        sa += mv * Wa[h * DD + d];
        se += mv * We[h * DD + d];
    }
#pragma unroll
    for (int off = 16; off; off >>= 1) {
        sa += __shfl_xor(sa, off, 32);
        se += __shfl_xor(se, off, 32);
    }
    if (lane == 0) {
        float abias = meta_raw[(size_t)b * NMETA + 1024 + h] + bm2[1024 + h];
        float ebias = meta_raw[(size_t)b * NMETA + 1032 + h] + bm2[1032 + h];
        alpha[b * HH + h] = sigmoidf_(sa + ba[h] + abias);
        eta[b * HH + h]   = sigmoidf_(se + be[h] + ebias) * 0.125f;  // * DH^-0.5
    }
}

// ---------------------------------------------------------------------------
// Streaming state update (HBM-bound, ~1 GB total):
//   k = l2norm(k_raw); err = v - S@k; new = (1-alpha)*S + eta*outer(err,k)
// One block per (b,h); 64x64 f32 tile staged in LDS; coalesced read & write.
// ---------------------------------------------------------------------------
__global__ void state_update(const float* __restrict__ old_state,
                             const float* __restrict__ k_raw,
                             const float* __restrict__ v_buf,
                             const float* __restrict__ alpha,
                             const float* __restrict__ eta,
                             float* __restrict__ out)
{
    __shared__ float sS[4096];
    __shared__ float kk[64];
    __shared__ float er[64];
    __shared__ float pm[256];
    const int t = threadIdx.x;
    const size_t bh = blockIdx.x;
    const float* sg = old_state + bh * 4096;
    for (int idx = t; idx < 4096; idx += 256) {
        sS[idx] = sg[idx];
        __builtin_prefetch(sg + idx + 4096, 0, 1);   // global_prefetch_b8 (next tile)
    }
    float kv = (t < 64) ? k_raw[bh * 64 + t] : 0.f;
    if (t < 64) pm[t] = kv * kv;
    __syncthreads();
    for (int s2 = 32; s2; s2 >>= 1) { if (t < s2) pm[t] += pm[t + s2]; __syncthreads(); }
    float inv = 1.f / fmaxf(sqrtf(pm[0]), 1e-12f);
    __syncthreads();
    if (t < 64) kk[t] = kv * inv;
    __syncthreads();
    {
        int i = t >> 2, seg = t & 3;
        float p = 0.f;
        const float* rowp = &sS[i * 64 + seg * 16];
        const float* kp = &kk[seg * 16];
#pragma unroll
        for (int j = 0; j < 16; ++j) p += rowp[j] * kp[j];
        pm[t] = p;
    }
    __syncthreads();
    if (t < 64) {
        float pred = pm[t * 4] + pm[t * 4 + 1] + pm[t * 4 + 2] + pm[t * 4 + 3];
        er[t] = v_buf[bh * 64 + t] - pred;
    }
    __syncthreads();
    const float al = alpha[bh], et = eta[bh];
    float* og = out + bh * 4096;
    for (int idx = t; idx < 4096; idx += 256) {
        int i = idx >> 6, j = idx & 63;
        og[idx] = (1.f - al) * sS[idx] + et * er[i] * kk[j];
    }
}

// ---------------------------------------------------------------------------
extern "C" void kernel_launch(void* const* d_in, const int* in_sizes, int n_in,
                              void* d_out, int out_size, void* d_ws, size_t ws_size,
                              hipStream_t stream)
{
    (void)in_sizes; (void)n_in; (void)out_size; (void)ws_size;
    const float* item      = (const float*)d_in[0];
    const float* old_state = (const float*)d_in[1];
    const float* user      = (const float*)d_in[2];
    const float* Wq  = (const float*)d_in[3];
    const float* Wk  = (const float*)d_in[4];
    const float* Wv  = (const float*)d_in[5];
    const float* Wa  = (const float*)d_in[6];
    const float* ba  = (const float*)d_in[7];
    const float* We  = (const float*)d_in[8];
    const float* be  = (const float*)d_in[9];
    const float* Wm1 = (const float*)d_in[10];
    const float* bm1 = (const float*)d_in[11];
    const float* gml = (const float*)d_in[12];
    const float* bml = (const float*)d_in[13];
    const float* Wm2 = (const float*)d_in[14];
    const float* bm2 = (const float*)d_in[15];
    const float* g1  = (const float*)d_in[16];
    const float* b1  = (const float*)d_in[17];
    float* out = (float*)d_out;

    char* p = (char*)d_ws;
    auto take = [&](size_t bytes) -> char* {
        char* r = p; p += (bytes + 255) & ~(size_t)255; return r;
    };
    float*    i_norm    = (float*)   take((size_t)BB * DD * 4);
    _Float16* i_norm_h  = (_Float16*)take((size_t)BB * DD * 2);
    _Float16* meta_in_h = (_Float16*)take((size_t)BB * 1024 * 2);
    float*    q_raw     = (float*)   take((size_t)BB * DD * 4);
    float*    hraw      = (float*)   take((size_t)BB * DD * 4);
    _Float16* hmid_h    = (_Float16*)take((size_t)BB * DD * 2);
    float*    meta_raw  = (float*)   take((size_t)BB * NMETA * 4);
    _Float16* mod_h     = (_Float16*)take((size_t)BB * DD * 2);
    float*    k_raw     = (float*)   take((size_t)BB * DD * 4);
    float*    v_buf     = (float*)   take((size_t)BB * DD * 4);
    float*    alpha     = (float*)   take((size_t)BB * HH * 4);
    float*    eta       = (float*)   take((size_t)BB * HH * 4);
    _Float16* Wq_h  = (_Float16*)take((size_t)DD * DD * 2);
    _Float16* Wk_h  = (_Float16*)take((size_t)DD * DD * 2);
    _Float16* Wv_h  = (_Float16*)take((size_t)DD * DD * 2);
    _Float16* Wm1_h = (_Float16*)take((size_t)DD * 1024 * 2);
    _Float16* Wm2_h = (_Float16*)take((size_t)NMETA * DD * 2);

    // 1) weight conversions to f16
    cvt_f16<<<DD * DD / 256, 256, 0, stream>>>(Wq, Wq_h, DD * DD);
    cvt_f16<<<DD * DD / 256, 256, 0, stream>>>(Wk, Wk_h, DD * DD);
    cvt_f16<<<DD * DD / 256, 256, 0, stream>>>(Wv, Wv_h, DD * DD);
    cvt_f16<<<DD * 1024 / 256, 256, 0, stream>>>(Wm1, Wm1_h, DD * 1024);
    cvt_f16<<<NMETA * DD / 256, 256, 0, stream>>>(Wm2, Wm2_h, NMETA * DD);

    // 2) layernorms (item + user_static)
    ln_dual<<<2 * BB, 256, 0, stream>>>(item, user, g1, b1, i_norm, i_norm_h, meta_in_h);

    // 3) q = i_norm @ Wq^T  (WMMA)
    wmma_gemm_nt<<<(BB / 16) * (DD / 16) / 8, 256, 0, stream>>>(i_norm_h, Wq_h, q_raw, BB, DD, DD);

    // 4) l2norm(q) + memory_context = S @ q  -> right half of meta_in
    qnorm_memctx<<<BB, 256, 0, stream>>>(q_raw, old_state, meta_in_h);

    // 5) hraw = meta_in @ Wm1^T (K=1024, WMMA), then LN+relu -> hmid
    wmma_gemm_nt<<<(BB / 16) * (DD / 16) / 8, 256, 0, stream>>>(meta_in_h, Wm1_h, hraw, BB, DD, 1024);
    ln_relu<<<BB, 256, 0, stream>>>(hraw, bm1, gml, bml, hmid_h);

    // 6) meta_out = hmid @ Wm2^T (N=1040, WMMA)
    wmma_gemm_nt<<<(BB / 16) * (NMETA / 16) / 8, 256, 0, stream>>>(hmid_h, Wm2_h, meta_raw, BB, NMETA, DD);

    // 7) modulated
    modulate<<<BB * DD / 256, 256, 0, stream>>>(meta_raw, bm2, i_norm, mod_h);

    // 8) k_raw, v (WMMA)
    wmma_gemm_nt<<<(BB / 16) * (DD / 16) / 8, 256, 0, stream>>>(mod_h, Wk_h, k_raw, BB, DD, DD);
    wmma_gemm_nt<<<(BB / 16) * (DD / 16) / 8, 256, 0, stream>>>(mod_h, Wv_h, v_buf, BB, DD, DD);

    // 9) gates
    ae_kernel<<<BB, 256, 0, stream>>>(mod_h, Wa, ba, We, be, meta_raw, bm2, alpha, eta);

    // 10) streaming state update -> d_out
    state_update<<<BB * HH, 256, 0, stream>>>(old_state, k_raw, v_buf, alpha, eta, out);
}
